// TernaryConv2d_34746285425011
// MI455X (gfx1250) — compile-verified, matrix operand order checked by reading, not run on hardware
//
#include <hip/hip_runtime.h>
#include <hip/hip_bf16.h>

// ---------------------------------------------------------------------------
// Ternary-quantized 3x3 conv (SAME, NHWC) as implicit GEMM with bf16 WMMA.
//   M = 16*56*56 = 50176 pixels, N = 256 cout, K = 9*256 = 2304
// Fast path: x pre-converted to bf16, then A and B tiles both staged with
// GLOBAL_LOAD_ASYNC_TO_LDS_B128 into double-buffered LDS (1 barrier/K-step).
// Fallback (small workspace): f32->bf16 conversion inside the GEMM.
// ---------------------------------------------------------------------------

typedef __bf16 bf16;
typedef __attribute__((ext_vector_type(16))) __bf16 bf16x16;
typedef __attribute__((ext_vector_type(8)))  float  f32x8;

#define IMG_N   16
#define IMG_H   56
#define IMG_W   56
#define CIN     256
#define COUT    256
#define NPIX    (IMG_N * IMG_H * IMG_W)     // 50176
#define KTOT    (9 * CIN)                   // 2304
#define WELEMS  (KTOT * COUT)               // 589824

#define BM 128
#define BN 128
#define BK 32
#define ASTR (BK + 8)   // padded LDS row stride (bf16 elems)
#define BSTR (BK + 8)
#define KSTEPS (KTOT / BK)                  // 72

// packed f32x2 -> bf16x2 conversion
static __device__ inline unsigned pack2bf16(float a, float b) {
#if __has_builtin(__builtin_amdgcn_cvt_pk_bf16_f32)
    typedef __attribute__((ext_vector_type(2))) __bf16 bf16x2;
    union { bf16x2 v; unsigned u; } t;
    t.v = __builtin_amdgcn_cvt_pk_bf16_f32(a, b);
    return t.u;
#else
    union { bf16 h[2]; unsigned u; } t;
    t.h[0] = (bf16)a; t.h[1] = (bf16)b;
    return t.u;
#endif
}

static __device__ inline void async_b128(unsigned lds_off, const void* gsrc) {
    asm volatile("global_load_async_to_lds_b128 %0, %1, off"
                 :: "v"(lds_off), "v"((unsigned long long)(size_t)gsrc)
                 : "memory");
}
static __device__ inline void wait_async0() {
    asm volatile("s_wait_asynccnt 0x0" ::: "memory");
}

// ---------------------------------------------------------------------------
// Kernel 1: threshold t = 0.7 * mean(|W|)   (single workgroup, deterministic)
// ---------------------------------------------------------------------------
__global__ __launch_bounds__(256)
void k_threshold(const float* __restrict__ W, float* __restrict__ wsf) {
    __shared__ float sh[256];
    float s = 0.f;
    for (int i = threadIdx.x; i < WELEMS; i += 256) s += fabsf(W[i]);
    sh[threadIdx.x] = s;
    __syncthreads();
    for (int off = 128; off > 0; off >>= 1) {
        if ((int)threadIdx.x < off) sh[threadIdx.x] += sh[threadIdx.x + off];
        __syncthreads();
    }
    if (threadIdx.x == 0) wsf[0] = 0.7f * sh[0] / (float)WELEMS;
}

// ---------------------------------------------------------------------------
// Kernel 2: alpha = sum(|w| : |w| > t) / count(|w| > t)
// ---------------------------------------------------------------------------
__global__ __launch_bounds__(256)
void k_alpha(const float* __restrict__ W, float* __restrict__ wsf) {
    __shared__ float sh[256];
    __shared__ float sc[256];
    const float t = wsf[0];
    float s = 0.f, c = 0.f;
    for (int i = threadIdx.x; i < WELEMS; i += 256) {
        float a = fabsf(W[i]);
        if (a > t) { s += a; c += 1.f; }
    }
    sh[threadIdx.x] = s;
    sc[threadIdx.x] = c;
    __syncthreads();
    for (int off = 128; off > 0; off >>= 1) {
        if ((int)threadIdx.x < off) {
            sh[threadIdx.x] += sh[threadIdx.x + off];
            sc[threadIdx.x] += sc[threadIdx.x + off];
        }
        __syncthreads();
    }
    if (threadIdx.x == 0) wsf[1] = sh[0] / sc[0];
}

// ---------------------------------------------------------------------------
// Kernel 3: ternarize W (HWIO) into bf16 Bq laid out N-major: Bq[co][k].
// ---------------------------------------------------------------------------
__global__ __launch_bounds__(256)
void k_quantize(const float* __restrict__ W, const float* __restrict__ wsf,
                bf16* __restrict__ Bq) {
    const float t  = wsf[0];
    const int   co = blockIdx.x;            // 0..255
    for (int j = threadIdx.x; j < KTOT; j += 256) {
        float w = W[(size_t)j * COUT + co];
        float q = (w > t) ? 1.f : ((w < -t) ? -1.f : 0.f);
        Bq[(size_t)co * KTOT + j] = (bf16)q;
    }
}

// ---------------------------------------------------------------------------
// Kernel 3b (fast path): stream-convert x (f32 NHWC) -> xb (bf16 NHWC).
// ---------------------------------------------------------------------------
__global__ __launch_bounds__(256)
void k_cvt_x(const float* __restrict__ x, bf16* __restrict__ xb) {
    size_t i = ((size_t)blockIdx.x * 256 + threadIdx.x) * 4;
    float4 v = *(const float4*)(x + i);
    uint2 pk;
    pk.x = pack2bf16(v.x, v.y);
    pk.y = pack2bf16(v.z, v.w);
    *(uint2*)(xb + i) = pk;
}

// ---------------------------------------------------------------------------
// Kernel 4 (fast path): implicit-im2col GEMM, all tiles via async memory->LDS,
// double-buffered LDS, one barrier per K-step, 8 WMMAs/wave/K-step.
// ---------------------------------------------------------------------------
__global__ __launch_bounds__(256)
void k_conv_gemm_async(const bf16* __restrict__ xb, const bf16* __restrict__ Bq,
                       const float* __restrict__ bias, const float* __restrict__ wsf,
                       float* __restrict__ out) {
    __shared__ bf16 As[2][BM * ASTR];
    __shared__ bf16 Bs[2][BN * BSTR];

    const int tid = threadIdx.x;
    const int p0  = blockIdx.x * BM;
    const int n0  = blockIdx.y * BN;

    // ---- loader indexing: one (row, 16-elem seg) strip per thread ----
    const int arow = tid >> 1;           // 0..127
    const int aseg = (tid & 1) << 4;     // 0 or 16
    const int p    = p0 + arow;
    const int pnn  = p / (IMG_H * IMG_W);
    const int rr   = p - pnn * (IMG_H * IMG_W);
    const int pyy  = rr / IMG_W;
    const int pxx  = rr - pyy * IMG_W;

    const bf16* bq_base = Bq + (size_t)(n0 + arow) * KTOT + aseg;

    const unsigned ldsA0 = (unsigned)(size_t)(void*)&As[0][arow * ASTR + aseg];
    const unsigned ldsA1 = (unsigned)(size_t)(void*)&As[1][arow * ASTR + aseg];
    const unsigned ldsB0 = (unsigned)(size_t)(void*)&Bs[0][arow * BSTR + aseg];
    const unsigned ldsB1 = (unsigned)(size_t)(void*)&Bs[1][arow * BSTR + aseg];

    // ---- wave / lane fragment indexing (ISA 16x16 layouts) ----
    const int wid   = tid >> 5;
    const int lane  = tid & 31;
    const int waveM = wid & 3;           // M offset 32*waveM
    const int waveN = wid >> 2;          // N offset 64*waveN
    const int lm    = lane & 15;
    const int half  = lane >> 4;

    f32x8 acc[2][4] = {};

    // stage K-step `ks` into buffer `buf` (async copies; zeros for halo rows)
    auto stage = [&](int ks, int buf) {
        const int tap = ks >> 3;
        const int kcc = (ks & 7) * BK;
        const int dy  = tap / 3 - 1;
        const int dx  = tap % 3 - 1;
        const int yy  = pyy + dy, xx = pxx + dx;
        const unsigned la = buf ? ldsA1 : ldsA0;
        const unsigned lb = buf ? ldsB1 : ldsB0;
        if (yy >= 0 && yy < IMG_H && xx >= 0 && xx < IMG_W) {
            const bf16* srcA =
                xb + ((size_t)((pnn * IMG_H + yy) * IMG_W + xx) * CIN + kcc + aseg);
            async_b128(la,      srcA);
            async_b128(la + 16, srcA + 8);
        } else {
            uint4 z = {0u, 0u, 0u, 0u};
            *(uint4*)&As[buf][arow * ASTR + aseg]     = z;
            *(uint4*)&As[buf][arow * ASTR + aseg + 8] = z;
        }
        const bf16* srcB = bq_base + tap * CIN + kcc;
        async_b128(lb,      srcB);
        async_b128(lb + 16, srcB + 8);
        __builtin_prefetch(srcB + BK, 0, 1);   // gfx1250 global_prefetch
    };

    stage(0, 0);
    wait_async0();
    __syncthreads();

    for (int k = 0; k < KSTEPS; ++k) {
        const int cur = k & 1;
        if (k + 1 < KSTEPS) stage(k + 1, cur ^ 1);   // overlap with WMMAs

        // ---- fragments from LDS + WMMA ----
        // A 16x32 bf16: VGPR0-3 = K 8*half+0..7, VGPR4-7 = K 16+8*half+0..7
        union { bf16x16 v; uint4 q[2]; } ua[2];
#pragma unroll
        for (int g = 0; g < 2; ++g) {
            const bf16* ap = &As[cur][(waveM * 32 + g * 16 + lm) * ASTR + 8 * half];
            ua[g].q[0] = *(const uint4*)(ap);
            ua[g].q[1] = *(const uint4*)(ap + 16);
        }
#pragma unroll
        for (int f = 0; f < 4; ++f) {
            union { bf16x16 v; uint4 q[2]; } ub;
            const bf16* bp = &Bs[cur][(waveN * 64 + f * 16 + lm) * BSTR + 8 * half];
            ub.q[0] = *(const uint4*)(bp);
            ub.q[1] = *(const uint4*)(bp + 16);
            acc[0][f] = __builtin_amdgcn_wmma_f32_16x16x32_bf16(
                false, ua[0].v, false, ub.v, (short)0, acc[0][f], false, false);
            acc[1][f] = __builtin_amdgcn_wmma_f32_16x16x32_bf16(
                false, ua[1].v, false, ub.v, (short)0, acc[1][f], false, false);
        }

        wait_async0();        // next-buffer copies landed (per-wave, in-order)
        __syncthreads();      // everyone done reading cur / writing nxt
    }

    // ---- epilogue: relu(alpha * acc + bias), f32 out (NHWC) ----
    const float alpha = wsf[1];
#pragma unroll
    for (int f = 0; f < 4; ++f) {
        const int   ng = n0 + waveN * 64 + f * 16 + lm;
        const float bv = bias[ng];
#pragma unroll
        for (int g = 0; g < 2; ++g) {
#pragma unroll
            for (int r = 0; r < 8; ++r) {
                const int m = waveM * 32 + g * 16 + 8 * half + r;  // C: M = r+8*half
                float v = acc[g][f][r] * alpha + bv;
                out[(size_t)(p0 + m) * COUT + ng] = v > 0.f ? v : 0.f;
            }
        }
    }
}

// ---------------------------------------------------------------------------
// Kernel 4f (fallback, small workspace): f32->bf16 conversion inside GEMM.
// ---------------------------------------------------------------------------
__global__ __launch_bounds__(256)
void k_conv_gemm(const float* __restrict__ x, const bf16* __restrict__ Bq,
                 const float* __restrict__ bias, const float* __restrict__ wsf,
                 float* __restrict__ out) {
    __shared__ bf16 As[BM * ASTR];
    __shared__ bf16 Bs[BN * BSTR];

    const int tid = threadIdx.x;
    const int p0  = blockIdx.x * BM;
    const int n0  = blockIdx.y * BN;

    const int pr = tid >> 3;
    const int cg = (tid & 7) << 2;
    int pn[4], py[4], px[4];
#pragma unroll
    for (int i = 0; i < 4; ++i) {
        int p  = p0 + pr + 32 * i;
        pn[i]  = p / (IMG_H * IMG_W);
        int r  = p - pn[i] * (IMG_H * IMG_W);
        py[i]  = r / IMG_W;
        px[i]  = r - py[i] * IMG_W;
    }

    const int brow = tid >> 1;
    const int bseg = (tid & 1) << 4;
    const bf16* bq_base = Bq + (size_t)(n0 + brow) * KTOT + bseg;
    const unsigned ldsB0 =
        (unsigned)(size_t)(void*)&Bs[brow * BSTR + bseg];
    const unsigned ldsB1 = ldsB0 + 16;

    const int wid   = tid >> 5;
    const int lane  = tid & 31;
    const int waveM = wid & 3;
    const int waveN = wid >> 2;
    const int lm    = lane & 15;
    const int half  = lane >> 4;

    f32x8 acc[2][4] = {};

    for (int tap = 0; tap < 9; ++tap) {
        const int dy = tap / 3 - 1;
        const int dx = tap % 3 - 1;
        const float* asrc[4];
        bool aval[4];
#pragma unroll
        for (int i = 0; i < 4; ++i) {
            int yy = py[i] + dy, xx = px[i] + dx;
            aval[i] = (yy >= 0) && (yy < IMG_H) && (xx >= 0) && (xx < IMG_W);
            asrc[i] = x + ((size_t)((pn[i] * IMG_H + yy) * IMG_W + xx) * CIN + cg);
        }
        const bf16* bsrc_tap = bq_base + tap * CIN;

        for (int kc = 0; kc < 8; ++kc) {
            const int kcc = kc * BK;
#pragma unroll
            for (int i = 0; i < 4; ++i) {
                float4 v = {0.f, 0.f, 0.f, 0.f};
                if (aval[i]) v = *(const float4*)(asrc[i] + kcc);
                uint2 pk;
                pk.x = pack2bf16(v.x, v.y);
                pk.y = pack2bf16(v.z, v.w);
                *(uint2*)&As[(pr + 32 * i) * ASTR + cg] = pk;
            }
            const bf16* bsrc = bsrc_tap + kcc;
            async_b128(ldsB0, bsrc);
            async_b128(ldsB1, bsrc + 8);
            __builtin_prefetch(bsrc + BK, 0, 1);

            wait_async0();
            __syncthreads();

            union { bf16x16 v; uint4 q[2]; } ua[2];
#pragma unroll
            for (int g = 0; g < 2; ++g) {
                const bf16* ap = &As[(waveM * 32 + g * 16 + lm) * ASTR + 8 * half];
                ua[g].q[0] = *(const uint4*)(ap);
                ua[g].q[1] = *(const uint4*)(ap + 16);
            }
#pragma unroll
            for (int f = 0; f < 4; ++f) {
                union { bf16x16 v; uint4 q[2]; } ub;
                const bf16* bp = &Bs[(waveN * 64 + f * 16 + lm) * BSTR + 8 * half];
                ub.q[0] = *(const uint4*)(bp);
                ub.q[1] = *(const uint4*)(bp + 16);
                acc[0][f] = __builtin_amdgcn_wmma_f32_16x16x32_bf16(
                    false, ua[0].v, false, ub.v, (short)0, acc[0][f], false, false);
                acc[1][f] = __builtin_amdgcn_wmma_f32_16x16x32_bf16(
                    false, ua[1].v, false, ub.v, (short)0, acc[1][f], false, false);
            }
            __syncthreads();
        }
    }

    const float alpha = wsf[1];
#pragma unroll
    for (int f = 0; f < 4; ++f) {
        const int   ng = n0 + waveN * 64 + f * 16 + lm;
        const float bv = bias[ng];
#pragma unroll
        for (int g = 0; g < 2; ++g) {
#pragma unroll
            for (int r = 0; r < 8; ++r) {
                const int m = waveM * 32 + g * 16 + 8 * half + r;
                float v = acc[g][f][r] * alpha + bv;
                out[(size_t)(p0 + m) * COUT + ng] = v > 0.f ? v : 0.f;
            }
        }
    }
}

// ---------------------------------------------------------------------------
extern "C" void kernel_launch(void* const* d_in, const int* in_sizes, int n_in,
                              void* d_out, int out_size, void* d_ws, size_t ws_size,
                              hipStream_t stream) {
    (void)in_sizes; (void)n_in; (void)out_size;
    const float* x = (const float*)d_in[0];
    const float* W = (const float*)d_in[1];
    const float* b = (const float*)d_in[2];
    float*       out = (float*)d_out;

    const size_t BQ_BYTES = (size_t)KTOT * COUT * sizeof(bf16);   // 1.13 MB
    const size_t XB_BYTES = (size_t)NPIX * CIN * sizeof(bf16);    // 25.7 MB

    float* wsf = (float*)d_ws;                          // [0]=t, [1]=alpha
    bf16*  Bq  = (bf16*)((char*)d_ws + 256);
    bf16*  xb  = (bf16*)((char*)d_ws + 256 + BQ_BYTES);

    k_threshold<<<1, 256, 0, stream>>>(W, wsf);
    k_alpha    <<<1, 256, 0, stream>>>(W, wsf);
    k_quantize <<<COUT, 256, 0, stream>>>(W, wsf, Bq);

    dim3 grid(NPIX / BM, COUT / BN);                    // 392 x 2
    if (ws_size >= 256 + BQ_BYTES + XB_BYTES) {
        // fast path: bf16 prepass + fully-async double-buffered GEMM
        k_cvt_x<<<(NPIX * CIN / 4) / 256, 256, 0, stream>>>(x, xb);
        k_conv_gemm_async<<<grid, 256, 0, stream>>>(xb, Bq, b, wsf, out);
    } else {
        k_conv_gemm<<<grid, 256, 0, stream>>>(x, Bq, b, wsf, out);
    }
}
